// Block_87428354277599
// MI455X (gfx1250) — matrix operand order repaired and checked by reading, not compiled, not run on hardware
//
#include <hip/hip_runtime.h>
#include <math.h>

// ---------------- types ----------------
typedef __attribute__((ext_vector_type(16))) __bf16 v16bf;
typedef __attribute__((ext_vector_type(8)))  float  f32x8;
typedef __attribute__((ext_vector_type(4)))  float  f32x4;

#define B_   2
#define T_   2048
#define C_   384
#define H_   6
#define HD_  64
#define CT_  385          // C+1 (time column)
#define KP_  416          // CT_ padded to multiple of 32
#define M_   (B_*T_)      // 4096 rows
#define BH_  (B_*H_)      // 12
#define LOG_T 7.6246189861f   // ln(2048)

// ---------------- WMMA fragment helpers (wave32, 16x16x32 bf16) ----------------
// A (16x32): lane L -> row M=L&15, holds K = kb..kb+7 and kb+16..kb+23, kb=(L>>4)*8
__device__ __forceinline__ v16bf load_frag_a(const __bf16* Arow, int lane, int k0) {
    int kr = k0 + ((lane >> 4) << 3);
    union { v16bf v; f32x4 f[2]; } u;
    u.f[0] = *reinterpret_cast<const f32x4*>(Arow + kr);
    u.f[1] = *reinterpret_cast<const f32x4*>(Arow + kr + 16);
    return u.v;
}
// B (32x16): lane L -> col N=n0+(L&15), holds K = kb..kb+7 and kb+16..kb+23, kb=k0+(L>>4)*8
__device__ __forceinline__ v16bf load_frag_b(const __bf16* Bm, int ldb, int lane, int k0, int n0) {
    int kr = k0 + ((lane >> 4) << 3);
    const __bf16* p = Bm + n0 + (lane & 15);
    v16bf v;
#pragma unroll
    for (int i = 0; i < 8; ++i) {
        v[i]     = p[(size_t)(kr + i) * ldb];
        v[i + 8] = p[(size_t)(kr + 16 + i) * ldb];
    }
    return v;
}
__device__ __forceinline__ f32x8 wmma_bf16(v16bf a, v16bf b, f32x8 c) {
    return __builtin_amdgcn_wmma_f32_16x16x32_bf16(false, a, false, b, (short)0, c, false, false);
}

// software-pipelined 16x64 strip GEMM over K: fragments for k-step n+1 are loaded
// before the WMMAs of k-step n are issued, so each v_wmma waits on loads that were
// issued a full iteration earlier (latency hidden behind matrix math).
__device__ __forceinline__ void gemm_strip(const __bf16* Arow, const __bf16* W,
                                           int K, int ldb, int n0, int lane,
                                           f32x8 acc[4]) {
    v16bf a  = load_frag_a(Arow, lane, 0);
    v16bf b0 = load_frag_b(W, ldb, lane, 0, n0);
    v16bf b1 = load_frag_b(W, ldb, lane, 0, n0 + 16);
    v16bf b2 = load_frag_b(W, ldb, lane, 0, n0 + 32);
    v16bf b3 = load_frag_b(W, ldb, lane, 0, n0 + 48);
    for (int k0 = 32; k0 < K; k0 += 32) {
        v16bf an = load_frag_a(Arow, lane, k0);
        v16bf c0 = load_frag_b(W, ldb, lane, k0, n0);
        v16bf c1 = load_frag_b(W, ldb, lane, k0, n0 + 16);
        v16bf c2 = load_frag_b(W, ldb, lane, k0, n0 + 32);
        v16bf c3 = load_frag_b(W, ldb, lane, k0, n0 + 48);
        acc[0] = wmma_bf16(a, b0, acc[0]);
        acc[1] = wmma_bf16(a, b1, acc[1]);
        acc[2] = wmma_bf16(a, b2, acc[2]);
        acc[3] = wmma_bf16(a, b3, acc[3]);
        a = an; b0 = c0; b1 = c1; b2 = c2; b3 = c3;
    }
    acc[0] = wmma_bf16(a, b0, acc[0]);
    acc[1] = wmma_bf16(a, b1, acc[1]);
    acc[2] = wmma_bf16(a, b2, acc[2]);
    acc[3] = wmma_bf16(a, b3, acc[3]);
}

// ---------------- block reductions (blockDim == 256) ----------------
__device__ __forceinline__ float block_sum(float v, float* sh) {
    int t = threadIdx.x; sh[t] = v; __syncthreads();
    for (int o = 128; o > 0; o >>= 1) { if (t < o) sh[t] += sh[t + o]; __syncthreads(); }
    float r = sh[0]; __syncthreads(); return r;
}
__device__ __forceinline__ float block_max(float v, float* sh) {
    int t = threadIdx.x; sh[t] = v; __syncthreads();
    for (int o = 128; o > 0; o >>= 1) { if (t < o) sh[t] = fmaxf(sh[t], sh[t + o]); __syncthreads(); }
    float r = sh[0]; __syncthreads(); return r;
}

// ---------------- small utility kernels ----------------
__global__ void zero_f32(float* p, int n) {
    int i = blockIdx.x * 256 + threadIdx.x;
    if (i < n) p[i] = 0.0f;
}
// weight fp32 (K x N) -> bf16 (Kpad x N), zero pad rows
__global__ void convert_pad(const float* src, __bf16* dst, int K, int N, int Kpad) {
    int idx = blockIdx.x * 256 + threadIdx.x;
    if (idx >= Kpad * N) return;
    int k = idx / N, n = idx - k * N;
    dst[idx] = (k < K) ? (__bf16)src[(size_t)k * N + n] : (__bf16)0.0f;
}
// add_time + LayerNorm over 385 -> bf16 padded row of 416
__global__ void addtime_ln(const float* xin, const float* tptr,
                           const float* w, const float* bb, __bf16* out) {
    int row = blockIdx.x;
    const float* x = xin + (size_t)row * C_;
    float tv = tptr[0];
    __shared__ float sh[256];
    float s = 0.f, s2 = 0.f;
    for (int j = threadIdx.x; j < CT_; j += 256) {
        float v = (j < C_) ? x[j] : tv;
        s += v; s2 += v * v;
    }
    float S  = block_sum(s,  sh);
    float S2 = block_sum(s2, sh);
    float mu  = S * (1.0f / CT_);
    float var = S2 * (1.0f / CT_) - mu * mu;
    float inv = rsqrtf(var + 1e-5f);
    __bf16* o = out + (size_t)row * KP_;
    for (int j = threadIdx.x; j < KP_; j += 256) {
        float v = 0.0f;
        if (j < CT_) {
            float xv = (j < C_) ? x[j] : tv;
            v = (xv - mu) * inv * w[j] + bb[j];
        }
        o[j] = (__bf16)v;
    }
}

// ---------------- GEMM kernels (wave computes 16x64 strip) ----------------
// qkv: xn(4096 x 416) @ W(416 x 1152) + bias -> scatter q (scaled 1/8), kT, v as bf16
__global__ void gemm_qkv(const __bf16* A, const __bf16* W, const float* bias,
                         __bf16* qb, __bf16* kTb, __bf16* vb) {
    int lane = threadIdx.x & 31, wave = threadIdx.x >> 5;
    int strip = blockIdx.x * 8 + wave;
    const int NS = (3 * C_) / 64;                 // 18
    int mt = strip / NS, nt = strip - mt * NS;
    if (mt >= M_ / 16) return;
    int m0 = mt << 4, n0 = nt << 6;
    const __bf16* Arow = A + (size_t)(m0 + (lane & 15)) * KP_;
    f32x8 acc[4] = {};
    gemm_strip(Arow, W, KP_, 3 * C_, n0, lane, acc);
    int nc = lane & 15, mb = (lane >> 4) << 3;
#pragma unroll
    for (int ss = 0; ss < 4; ++ss) {
        int j = n0 + ss * 16 + nc;
        float bj = bias[j];
        int g = j / C_, rem = j - g * C_;
        int h = rem >> 6, d = rem & 63;
#pragma unroll
        for (int i = 0; i < 8; ++i) {
            int row = m0 + mb + i;
            int bidx = row >> 11, t = row & (T_ - 1);
            size_t bh = (size_t)(bidx * H_ + h);
            float val = acc[ss][i] + bj;
            if (g == 0)      qb [(bh * T_ + t) * HD_ + d] = (__bf16)(val * 0.125f);
            else if (g == 1) kTb[(bh * HD_ + d) * T_ + t] = (__bf16)val;
            else             vb [(bh * T_ + t) * HD_ + d] = (__bf16)val;
        }
    }
}

// logits: per (b,h): q(2048x64) @ kT(64x2048) -> att f32; skip strictly-upper strips
__global__ void gemm_logits(const __bf16* qb, const __bf16* kTb, float* att) {
    int lane = threadIdx.x & 31, wave = threadIdx.x >> 5;
    int strip = blockIdx.x * 8 + wave;
    int mt = strip >> 5, nt = strip & 31;        // 128 x 32
    if (mt >= T_ / 16) return;
    int m0 = mt << 4, n0 = nt << 6;
    if (n0 > m0 + 15) return;                    // above causal diagonal: never read
    size_t bh = blockIdx.y;
    const __bf16* q  = qb  + bh * (size_t)T_ * HD_;
    const __bf16* kT = kTb + bh * (size_t)HD_ * T_;
    float* out = att + bh * (size_t)T_ * T_;
    const __bf16* Arow = q + (size_t)(m0 + (lane & 15)) * HD_;
    f32x8 acc[4] = {};
    gemm_strip(Arow, kT, HD_, T_, n0, lane, acc);
    int nc = lane & 15, mb = (lane >> 4) << 3;
#pragma unroll
    for (int ss = 0; ss < 4; ++ss)
#pragma unroll
        for (int i = 0; i < 8; ++i)
            out[(size_t)(m0 + mb + i) * T_ + n0 + ss * 16 + nc] = acc[ss][i];
}

// causal row softmax in place; writes exact zeros for j > t (reference semantics)
__global__ void softmax_causal(float* att) {
    int t = blockIdx.x; size_t bh = blockIdx.y;
    float* row = att + (bh * T_ + (size_t)t) * T_;
    __shared__ float sh[256];
    float m = -1e30f;
    for (int j = threadIdx.x; j <= t; j += 256) m = fmaxf(m, row[j]);
    m = block_max(m, sh);
    float s = 0.f;
    for (int j = threadIdx.x; j <= t; j += 256) s += __expf(row[j] - m);
    s = block_sum(s, sh);
    float inv = 1.0f / s;
    for (int j = threadIdx.x; j < T_; j += 256)
        row[j] = (j <= t) ? __expf(row[j] - m) * inv : 0.0f;
}

// sinkhorn u-update: u_i = -logT - lse_j(c_ij + v_j)
// single pass over the row: per-thread streaming LSE, then (m,s)-pair tree combine
__global__ void sink_row(const float* att, const float* vvec, float* u) {
    int t = blockIdx.x; size_t bh = blockIdx.y;
    const float* row = att + (bh * T_ + (size_t)t) * T_;
    const float* vv  = vvec + bh * T_;
    __shared__ float shm[256], shs[256];
    float m = -1e30f, s = 0.f;
    for (int j = threadIdx.x; j < T_; j += 256) {
        float x = row[j] + vv[j];
        if (x > m) { s = s * __expf(m - x) + 1.0f; m = x; }
        else       { s += __expf(x - m); }
    }
    int tid = threadIdx.x;
    shm[tid] = m; shs[tid] = s; __syncthreads();
    for (int o = 128; o > 0; o >>= 1) {
        if (tid < o) {
            float m1 = shm[tid], s1 = shs[tid];
            float m2 = shm[tid + o], s2 = shs[tid + o];
            float mm = fmaxf(m1, m2);
            shs[tid] = s1 * __expf(m1 - mm) + s2 * __expf(m2 - mm);
            shm[tid] = mm;
        }
        __syncthreads();
    }
    if (tid == 0) u[bh * T_ + t] = -LOG_T - (shm[0] + __logf(shs[0]));
}

// sinkhorn v-update: v_j = -logT - lse_i(c_ij + u_i); one thread per column, streaming LSE
__global__ void sink_col(const float* att, const float* u, float* vvec) {
    int j = blockIdx.x * 256 + threadIdx.x;
    size_t bh = blockIdx.y;
    const float* c  = att + bh * (size_t)T_ * T_;
    const float* uu = u + bh * T_;
    float m = -1e30f, s = 0.f;
    for (int i = 0; i < T_; ++i) {
        float x = c[(size_t)i * T_ + j] + uu[i];
        if (x > m) { s = s * __expf(m - x) + 1.0f; m = x; }
        else       { s += __expf(x - m); }
    }
    vvec[bh * T_ + j] = -LOG_T - (m + __logf(s));
}

// y = (exp(c+u+v)*T) @ v ; pi built per-fragment; output scattered to (B,T,384) bf16
// B-fragment loads are issued first each k-step; the 16 exp() VALU ops that build the
// A-fragment overlap the B-load latency.
__global__ void gemm_pi_v(const float* att, const float* u, const float* vvec,
                          const __bf16* vb, __bf16* ybuf) {
    int lane = threadIdx.x & 31, wave = threadIdx.x >> 5;
    int mt = blockIdx.x * 8 + wave;
    if (mt >= T_ / 16) return;
    size_t bh = blockIdx.y;
    int bidx = (int)(bh / H_), h = (int)(bh % H_);
    int m0 = mt << 4;
    int mrow = m0 + (lane & 15);
    const float* crow = att + (bh * T_ + (size_t)mrow) * T_;
    const float* vv = vvec + bh * T_;
    float um = u[bh * T_ + mrow];
    const __bf16* Bv = vb + bh * (size_t)T_ * HD_;
    f32x8 acc[4] = {};
    for (int k0 = 0; k0 < T_; k0 += 32) {
        v16bf b0 = load_frag_b(Bv, HD_, lane, k0, 0);
        v16bf b1 = load_frag_b(Bv, HD_, lane, k0, 16);
        v16bf b2 = load_frag_b(Bv, HD_, lane, k0, 32);
        v16bf b3 = load_frag_b(Bv, HD_, lane, k0, 48);
        int kr = k0 + ((lane >> 4) << 3);
        union { f32x4 v4[4]; float f[16]; } cu, vu;
        cu.v4[0] = *reinterpret_cast<const f32x4*>(crow + kr);
        cu.v4[1] = *reinterpret_cast<const f32x4*>(crow + kr + 4);
        cu.v4[2] = *reinterpret_cast<const f32x4*>(crow + kr + 16);
        cu.v4[3] = *reinterpret_cast<const f32x4*>(crow + kr + 20);
        vu.v4[0] = *reinterpret_cast<const f32x4*>(vv + kr);
        vu.v4[1] = *reinterpret_cast<const f32x4*>(vv + kr + 4);
        vu.v4[2] = *reinterpret_cast<const f32x4*>(vv + kr + 16);
        vu.v4[3] = *reinterpret_cast<const f32x4*>(vv + kr + 20);
        v16bf a;
#pragma unroll
        for (int i = 0; i < 16; ++i)
            a[i] = (__bf16)(__expf(cu.f[i] + um + vu.f[i]) * (float)T_);
        acc[0] = wmma_bf16(a, b0, acc[0]);
        acc[1] = wmma_bf16(a, b1, acc[1]);
        acc[2] = wmma_bf16(a, b2, acc[2]);
        acc[3] = wmma_bf16(a, b3, acc[3]);
    }
    int nc = lane & 15, mb = (lane >> 4) << 3;
#pragma unroll
    for (int ss = 0; ss < 4; ++ss)
#pragma unroll
        for (int i = 0; i < 8; ++i) {
            int t = m0 + mb + i;
            ybuf[((size_t)bidx * T_ + t) * C_ + h * HD_ + ss * 16 + nc] = (__bf16)acc[ss][i];
        }
}

// generic: A(M x K bf16, lda=K) @ W(K x N bf16) + bias -> f32 out (M x N)
__global__ void gemm_bias_f32(const __bf16* A, const __bf16* W, const float* bias,
                              float* out, int M, int K, int N) {
    int lane = threadIdx.x & 31, wave = threadIdx.x >> 5;
    int strip = blockIdx.x * 8 + wave;
    int ns = N >> 6;
    int mt = strip / ns, nt = strip - mt * ns;
    if (mt >= (M >> 4)) return;
    int m0 = mt << 4, n0 = nt << 6;
    const __bf16* Arow = A + (size_t)(m0 + (lane & 15)) * K;
    f32x8 acc[4] = {};
    gemm_strip(Arow, W, K, N, n0, lane, acc);
    int nc = lane & 15, mb = (lane >> 4) << 3;
#pragma unroll
    for (int ss = 0; ss < 4; ++ss) {
        float bj = bias[n0 + ss * 16 + nc];
#pragma unroll
        for (int i = 0; i < 8; ++i)
            out[(size_t)(m0 + mb + i) * N + n0 + ss * 16 + nc] = acc[ss][i] + bj;
    }
}

// generic with exact GELU epilogue -> bf16 out
__global__ void gemm_bias_gelu_bf16(const __bf16* A, const __bf16* W, const float* bias,
                                    __bf16* out, int M, int K, int N) {
    int lane = threadIdx.x & 31, wave = threadIdx.x >> 5;
    int strip = blockIdx.x * 8 + wave;
    int ns = N >> 6;
    int mt = strip / ns, nt = strip - mt * ns;
    if (mt >= (M >> 4)) return;
    int m0 = mt << 4, n0 = nt << 6;
    const __bf16* Arow = A + (size_t)(m0 + (lane & 15)) * K;
    f32x8 acc[4] = {};
    gemm_strip(Arow, W, K, N, n0, lane, acc);
    int nc = lane & 15, mb = (lane >> 4) << 3;
#pragma unroll
    for (int ss = 0; ss < 4; ++ss) {
        float bj = bias[n0 + ss * 16 + nc];
#pragma unroll
        for (int i = 0; i < 8; ++i) {
            float v = acc[ss][i] + bj;
            float g = 0.5f * v * (1.0f + erff(v * 0.70710678118f));
            out[(size_t)(m0 + mb + i) * N + n0 + ss * 16 + nc] = (__bf16)g;
        }
    }
}

// ---------------- host ----------------
extern "C" void kernel_launch(void* const* d_in, const int* in_sizes, int n_in,
                              void* d_out, int out_size, void* d_ws, size_t ws_size,
                              hipStream_t stream) {
    const float* x      = (const float*)d_in[0];
    const float* t      = (const float*)d_in[1];
    const float* ln1_w  = (const float*)d_in[2];
    const float* ln1_b  = (const float*)d_in[3];
    const float* attn_w = (const float*)d_in[4];
    const float* attn_b = (const float*)d_in[5];
    const float* proj_w = (const float*)d_in[6];
    const float* proj_b = (const float*)d_in[7];
    const float* ln2_w  = (const float*)d_in[8];
    const float* ln2_b  = (const float*)d_in[9];
    const float* fc_w   = (const float*)d_in[10];
    const float* fc_b   = (const float*)d_in[11];
    const float* fc2_w  = (const float*)d_in[12];
    const float* fc2_b  = (const float*)d_in[13];
    float* out = (float*)d_out;

    char* ws = (char*)d_ws;
    size_t off = 0;
    auto alloc = [&](size_t bytes) { char* p = ws + off; off += (bytes + 255) & ~(size_t)255; return p; };

    float*  att      = (float*) alloc((size_t)BH_ * T_ * T_ * 4);  // 201 MB
    __bf16* xn       = (__bf16*)alloc((size_t)M_ * KP_ * 2);
    __bf16* attn_wb  = (__bf16*)alloc((size_t)KP_ * 3 * C_ * 2);
    __bf16* proj_wb  = (__bf16*)alloc((size_t)C_ * C_ * 2);
    __bf16* fc_wb    = (__bf16*)alloc((size_t)KP_ * 4 * C_ * 2);
    __bf16* fc2_wb   = (__bf16*)alloc((size_t)4 * C_ * C_ * 2);
    __bf16* qb       = (__bf16*)alloc((size_t)BH_ * T_ * HD_ * 2);
    __bf16* kTb      = (__bf16*)alloc((size_t)BH_ * HD_ * T_ * 2);
    __bf16* vb       = (__bf16*)alloc((size_t)BH_ * T_ * HD_ * 2);
    float*  uvec     = (float*) alloc((size_t)BH_ * T_ * 4);
    float*  vvec     = (float*) alloc((size_t)BH_ * T_ * 4);
    __bf16* ybuf     = (__bf16*)alloc((size_t)M_ * C_ * 2);
    float*  hbuf     = (float*) alloc((size_t)M_ * C_ * 4);
    __bf16* hn       = (__bf16*)alloc((size_t)M_ * KP_ * 2);
    __bf16* mbuf     = (__bf16*)alloc((size_t)M_ * 4 * C_ * 2);
    (void)ws_size; (void)n_in; (void)in_sizes; (void)out_size;

    // weight conversions (fp32 -> bf16, K-padded)
    convert_pad<<<(KP_ * 3 * C_ + 255) / 256, 256, 0, stream>>>(attn_w, attn_wb, CT_, 3 * C_, KP_);
    convert_pad<<<(C_ * C_ + 255) / 256, 256, 0, stream>>>(proj_w, proj_wb, C_, C_, C_);
    convert_pad<<<(KP_ * 4 * C_ + 255) / 256, 256, 0, stream>>>(fc_w, fc_wb, CT_, 4 * C_, KP_);
    convert_pad<<<(4 * C_ * C_ + 255) / 256, 256, 0, stream>>>(fc2_w, fc2_wb, 4 * C_, C_, 4 * C_);

    // ---- attention sub-block ----
    addtime_ln<<<M_, 256, 0, stream>>>(x, t, ln1_w, ln1_b, xn);
    gemm_qkv<<<(M_ / 16) * ((3 * C_) / 64) / 8, 256, 0, stream>>>(xn, attn_wb, attn_b, qb, kTb, vb);
    gemm_logits<<<dim3((T_ / 16) * (T_ / 64) / 8, BH_), 256, 0, stream>>>(qb, kTb, att);
    softmax_causal<<<dim3(T_, BH_), 256, 0, stream>>>(att);

    // ---- sinkhorn: 3 x (row update, col update) ----
    zero_f32<<<(BH_ * T_ + 255) / 256, 256, 0, stream>>>(vvec, BH_ * T_);
    for (int it = 0; it < 3; ++it) {
        sink_row<<<dim3(T_, BH_), 256, 0, stream>>>(att, vvec, uvec);
        sink_col<<<dim3(T_ / 256, BH_), 256, 0, stream>>>(att, uvec, vvec);
    }

    // ---- y = pi @ v, proj ----
    gemm_pi_v<<<dim3((T_ / 16) / 8, BH_), 256, 0, stream>>>(att, uvec, vvec, vb, ybuf);
    gemm_bias_f32<<<(M_ / 16) * (C_ / 64) / 8, 256, 0, stream>>>(ybuf, proj_wb, proj_b, hbuf, M_, C_, C_);

    // ---- MLP sub-block ----
    addtime_ln<<<M_, 256, 0, stream>>>(hbuf, t, ln2_w, ln2_b, hn);
    gemm_bias_gelu_bf16<<<(M_ / 16) * ((4 * C_) / 64) / 8, 256, 0, stream>>>(hn, fc_wb, fc_b, mbuf, M_, KP_, 4 * C_);
    gemm_bias_f32<<<(M_ / 16) * (C_ / 64) / 8, 256, 0, stream>>>(mbuf, fc2_wb, fc2_b, out, M_, 4 * C_, C_);
}
